// HoT_86431921864798
// MI455X (gfx1250) — compile-verified, matrix operand order checked by reading, not run on hardware
//
#include <hip/hip_runtime.h>
#include <math.h>

// ---------------- model constants ----------------
#define BZ   8
#define FF   243
#define JJ   17
#define DM   256
#define HN   8
#define HDm  32
#define HID  1024
#define SEQ2 81
#define DREP 512
#define CDIV(a,b) (((a)+(b)-1)/(b))

typedef __attribute__((ext_vector_type(16))) __bf16 v16bf;
typedef __attribute__((ext_vector_type(8)))  __bf16 v8bf;
typedef __attribute__((ext_vector_type(8)))  float  v8f;

// ---------------- weight prep: W[K,N] fp32 -> Wt[N,K] bf16 ----------------
__global__ __launch_bounds__(256) void wprep_k(
    const float* __restrict__ W, __bf16* __restrict__ Wt, int K, int N)
{
    int t = blockIdx.x * blockDim.x + threadIdx.x;
    if (t >= K * N) return;
    const int n = t / K, k = t - n * K;
    Wt[t] = (__bf16)W[(size_t)k * N + n];
}

// =================================================================
// WMMA GEMM:  C[M,N] = act(A[M,K] @ B[K,N] + bias) (+ res)
// A fp32 row-major (cvt to bf16 inline), Bt = bf16 transposed [N,K].
// One wave computes a 16 x (16*NT) tile via v_wmma_f32_16x16x32_bf16.
// act: 0 none, 1 exact GELU, 2 tanh.
// =================================================================
template<int NT>
__global__ __launch_bounds__(256) void wmma_gemm_k(
    const float* __restrict__ A, const __bf16* __restrict__ Bt,
    const float* __restrict__ bias, const float* res,
    float* C, int M, int N, int K, int act)
{
    const int lane = threadIdx.x & 31;
    const int wave = threadIdx.x >> 5;
    const int tilesN = (N + 16*NT - 1) / (16*NT);
    const int tilesM = (M + 15) >> 4;
    const int tile = blockIdx.x * (blockDim.x >> 5) + wave;
    if (tile >= tilesM * tilesN) return;
    const int tm = tile / tilesN, tn = tile - tm * tilesN;
    const int m0 = tm << 4, n0 = tn * 16 * NT;
    const int lm = lane & 15, lg = lane >> 4;

    v8f acc[NT];
#pragma unroll
    for (int t = 0; t < NT; ++t)
#pragma unroll
        for (int r = 0; r < 8; ++r) acc[t][r] = 0.0f;

    // Out-of-range A rows / B cols only feed D rows/cols that are never
    // stored, so clamp the pointers instead of branching in the hot loop.
    const int am = m0 + lm;
    const float* Arow = A + (size_t)(am < M ? am : 0) * K;
    const __bf16* Btn[NT];
#pragma unroll
    for (int t = 0; t < NT; ++t) {
        const int n = n0 + t*16 + lm;
        Btn[t] = Bt + (size_t)(n < N ? n : 0) * K;
    }

    for (int k0 = 0; k0 < K; k0 += 32) {
        const int kb = k0 + lg * 8;
        const float4 a0 = *(const float4*)(Arow + kb);
        const float4 a1 = *(const float4*)(Arow + kb + 4);
        const float4 a2 = *(const float4*)(Arow + kb + 16);
        const float4 a3 = *(const float4*)(Arow + kb + 20);
        if (k0 + 32 < K) __builtin_prefetch(Arow + kb + 32, 0, 1);
        v16bf af;
        af[0]=(__bf16)a0.x; af[1]=(__bf16)a0.y; af[2]=(__bf16)a0.z; af[3]=(__bf16)a0.w;
        af[4]=(__bf16)a1.x; af[5]=(__bf16)a1.y; af[6]=(__bf16)a1.z; af[7]=(__bf16)a1.w;
        af[8]=(__bf16)a2.x; af[9]=(__bf16)a2.y; af[10]=(__bf16)a2.z; af[11]=(__bf16)a2.w;
        af[12]=(__bf16)a3.x; af[13]=(__bf16)a3.y; af[14]=(__bf16)a3.z; af[15]=(__bf16)a3.w;

#pragma unroll
        for (int t = 0; t < NT; ++t) {
            const v8bf b0 = *(const v8bf*)(Btn[t] + kb);        // K = kb..kb+7
            const v8bf b1 = *(const v8bf*)(Btn[t] + kb + 16);   // K = kb+16..kb+23
            const v16bf bfv = __builtin_shufflevector(b0, b1,
                0,1,2,3,4,5,6,7,8,9,10,11,12,13,14,15);
            acc[t] = __builtin_amdgcn_wmma_f32_16x16x32_bf16(
                false, af, false, bfv, (short)0, acc[t], false, false);
        }
    }

#pragma unroll
    for (int t = 0; t < NT; ++t) {
        const int n = n0 + t*16 + lm;
        if (n >= N) continue;
        const float bv = bias ? bias[n] : 0.0f;
#pragma unroll
        for (int r = 0; r < 8; ++r) {
            const int m = m0 + r + lg * 8;
            if (m >= M) continue;
            float v = acc[t][r] + bv;
            if (act == 1)      v = 0.5f * v * (1.0f + erff(v * 0.70710678118654752f));
            else if (act == 2) v = tanhf(v);
            if (res) v += res[(size_t)m * N + n];
            C[(size_t)m * N + n] = v;
        }
    }
}

// ---------------- LayerNorm over 256, one wave per row ----------------
__global__ __launch_bounds__(256) void ln256_k(
    const float* __restrict__ x, const float* __restrict__ w,
    const float* __restrict__ b, float* __restrict__ y, int rows)
{
    const int gw   = (blockIdx.x * blockDim.x + threadIdx.x) >> 5;
    const int lane = threadIdx.x & 31;
    if (gw >= rows) return;
    const float* xr = x + (size_t)gw * DM;
    float4 v0 = *(const float4*)(xr + lane * 8);
    float4 v1 = *(const float4*)(xr + lane * 8 + 4);
    float s = v0.x+v0.y+v0.z+v0.w + v1.x+v1.y+v1.z+v1.w;
    for (int m = 16; m >= 1; m >>= 1) s += __shfl_xor(s, m, 32);
    const float mu = s * (1.0f / 256.0f);
    float d[8] = { v0.x-mu, v0.y-mu, v0.z-mu, v0.w-mu, v1.x-mu, v1.y-mu, v1.z-mu, v1.w-mu };
    float ss = 0.f;
#pragma unroll
    for (int i = 0; i < 8; ++i) ss += d[i]*d[i];
    for (int m = 16; m >= 1; m >>= 1) ss += __shfl_xor(ss, m, 32);
    const float rs = rsqrtf(ss * (1.0f / 256.0f) + 1e-6f);
    const int c = lane * 8;
    float* yr = y + (size_t)gw * DM;
#pragma unroll
    for (int i = 0; i < 8; ++i) yr[c+i] = d[i] * rs * w[c+i] + b[c+i];
}

// ---------------- fused online-softmax attention (HD=32) ----------------
__device__ __forceinline__ void attn_one_row(
    const float* __restrict__ q, const float* __restrict__ kb,
    const float* __restrict__ vb, int nkv, int stride, float* out)
{
    float qr[32];
#pragma unroll
    for (int d = 0; d < 32; ++d) qr[d] = q[d];
    float mx = -1e30f, l = 0.0f, acc[32];
#pragma unroll
    for (int d = 0; d < 32; ++d) acc[d] = 0.0f;
    for (int j = 0; j < nkv; ++j) {
        const float* kr = kb + (size_t)j * stride;
        float s = 0.0f;
#pragma unroll
        for (int d = 0; d < 32; ++d) s += qr[d] * kr[d];
        s *= 0.17677669529663687f;                 // HD^-0.5
        const float nm = fmaxf(mx, s);
        const float corr = __expf(mx - nm);
        const float p    = __expf(s - nm);
        l = l * corr + p;
        const float* vr = vb + (size_t)j * stride;
#pragma unroll
        for (int d = 0; d < 32; ++d) acc[d] = acc[d] * corr + p * vr[d];
        mx = nm;
    }
    const float inv = 1.0f / l;
#pragma unroll
    for (int d = 0; d < 32; ++d) out[d] = acc[d] * inv;
}

__global__ __launch_bounds__(128) void attn_sp_k(const float* qkv, float* out, int BN)
{
    int t = blockIdx.x * blockDim.x + threadIdx.x;
    if (t >= BN * HN * JJ) return;
    const int n = t % JJ; int rem = t / JJ;
    const int h = rem & 7; const int bn = rem >> 3;
    const size_t qrow = (size_t)bn * JJ + n;
    const float* kb = qkv + ((size_t)bn * JJ) * 768 + 256 + h * 32;
    attn_one_row(qkv + qrow * 768 + h * 32, kb, kb + 256, JJ, 768,
                 out + qrow * DM + h * 32);
}

__global__ __launch_bounds__(128) void attn_tm_k(const float* qkv, float* out, int seq)
{
    int t = blockIdx.x * blockDim.x + threadIdx.x;
    if (t >= BZ * HN * JJ * seq) return;
    const int f = t % seq; int rem = t / seq;
    const int j = rem % JJ; rem /= JJ;
    const int h = rem & 7; const int b = rem >> 3;
    const size_t qrow = ((size_t)(b * seq + f)) * JJ + j;
    const float* kb = qkv + (((size_t)b * seq) * JJ + j) * 768 + 256 + h * 32;
    attn_one_row(qkv + qrow * 768 + h * 32, kb, kb + 256, seq, JJ * 768,
                 out + qrow * DM + h * 32);
}

__global__ __launch_bounds__(128) void attn_cr_k(const float* qsm, const float* kbuf,
                                                 const float* vbuf, float* out)
{
    int t = blockIdx.x * blockDim.x + threadIdx.x;
    if (t >= BZ * JJ * HN * FF) return;
    const int f = t % FF; int rem = t / FF;
    const int h = rem & 7; const int bj = rem >> 3;
    attn_one_row(qsm + (size_t)f * DM + h * 32,
                 kbuf + (size_t)bj * SEQ2 * DM + h * 32,
                 vbuf + (size_t)bj * SEQ2 * DM + h * 32,
                 SEQ2, DM, out + ((size_t)bj * FF + f) * DM + h * 32);
}

// ---------------- embedding: x@W3x256 + biases + pos/temp embeds ----------------
__global__ __launch_bounds__(256) void embed_k(
    const float* __restrict__ x, const float* __restrict__ W,
    const float* __restrict__ bias, const float* __restrict__ pos,
    const float* __restrict__ temp, float* __restrict__ h, int total)
{
    int t = blockIdx.x * blockDim.x + threadIdx.x;
    if (t >= total) return;
    const int c = t & 255; const int row = t >> 8;
    const int j = row % JJ; const int f = (row / JJ) % FF;
    const float* xr = x + (size_t)row * 3;
    h[t] = xr[0]*W[c] + xr[1]*W[256+c] + xr[2]*W[512+c]
         + bias[c] + pos[j*DM + c] + temp[f*DM + c];
}

// ---------------- DPC-KNN pieces ----------------
__global__ __launch_bounds__(256) void fmean_k(const float* h, float* hf)
{
    int t = blockIdx.x * blockDim.x + threadIdx.x;
    if (t >= BZ * FF * DM) return;
    const int c = t & 255; const int bf = t >> 8;
    const float* base = h + ((size_t)bf * JJ) * DM + c;
    float s = 0.f;
#pragma unroll
    for (int j = 0; j < JJ; ++j) s += base[(size_t)j * DM];
    hf[t] = s * (1.0f / JJ);
}

__global__ __launch_bounds__(256) void dist_k(const float* hf, float* dist)
{
    int t = blockIdx.x * blockDim.x + threadIdx.x;
    if (t >= BZ * FF * FF) return;
    const int j = t % FF; const int i = (t / FF) % FF; const int b = t / (FF * FF);
    const float* xi = hf + ((size_t)b * FF + i) * DM;
    const float* xj = hf + ((size_t)b * FF + j) * DM;
    float acc = 0.f;
    for (int c = 0; c < DM; c += 4) {
        float4 a = *(const float4*)(xi + c), bb = *(const float4*)(xj + c);
        float dx=a.x-bb.x, dy=a.y-bb.y, dz=a.z-bb.z, dw=a.w-bb.w;
        acc += dx*dx + dy*dy + dz*dz + dw*dw;
    }
    dist[t] = sqrtf(acc) * (1.0f / 16.0f);    // / sqrt(DM)
}

__global__ void dmax_k(const float* dist, float* dmax)
{
    __shared__ float sm[256];
    const int b = blockIdx.x;
    const float* d = dist + (size_t)b * FF * FF;
    float mx = 0.f;
    for (int i = threadIdx.x; i < FF * FF; i += 256) mx = fmaxf(mx, d[i]);
    sm[threadIdx.x] = mx; __syncthreads();
    for (int s = 128; s > 0; s >>= 1) {
        if (threadIdx.x < s) sm[threadIdx.x] = fmaxf(sm[threadIdx.x], sm[threadIdx.x + s]);
        __syncthreads();
    }
    if (threadIdx.x == 0) dmax[b] = sm[0];
}

__global__ __launch_bounds__(256) void density_k(const float* dist, float* dens)
{
    int t = blockIdx.x * blockDim.x + threadIdx.x;
    if (t >= BZ * FF) return;
    const int b = t / FF, i = t % FF;
    const float* dr = dist + ((size_t)b * FF + i) * FF;
    float best[5] = {1e30f,1e30f,1e30f,1e30f,1e30f};
    for (int j = 0; j < FF; ++j) {
        const float d = dr[j];
        int mi = 0; float mv = best[0];
#pragma unroll
        for (int q = 1; q < 5; ++q) if (best[q] > mv) { mv = best[q]; mi = q; }
        if (d < mv) best[mi] = d;
    }
    float s = 0.f;
#pragma unroll
    for (int q = 0; q < 5; ++q) s += best[q] * best[q];
    float den = __expf(-s * 0.2f);
    unsigned int hsh = (unsigned)(i * 1664525u) ^ (unsigned)(b * 1013904223u);
    hsh *= 2654435761u; hsh ^= hsh >> 16; hsh *= 2246822519u;
    den += (float)(hsh >> 8) * (1.0f / 16777216.0f) * 1e-6f;   // deterministic tie-break
    dens[t] = den;
}

__global__ __launch_bounds__(256) void score_k(const float* dist, const float* dens,
                                               const float* dmax, float* score)
{
    int t = blockIdx.x * blockDim.x + threadIdx.x;
    if (t >= BZ * FF) return;
    const int b = t / FF, i = t % FF;
    const float di = dens[(size_t)b * FF + i];
    const float* dr = dist + ((size_t)b * FF + i) * FF;
    float mn = dmax[b];
    for (int j = 0; j < FF; ++j)
        if (dens[(size_t)b * FF + j] > di) mn = fminf(mn, dr[j]);
    score[t] = mn * di;
}

__global__ void select81_k(const float* score, int* idx)
{
    const int b = blockIdx.x;
    if (threadIdx.x != 0) return;
    float sc[FF];
    for (int i = 0; i < FF; ++i) sc[i] = score[(size_t)b * FF + i];
    int ch[SEQ2];
    for (int s = 0; s < SEQ2; ++s) {
        int am = 0; float mv = sc[0];
        for (int i = 1; i < FF; ++i) if (sc[i] > mv) { mv = sc[i]; am = i; }
        ch[s] = am; sc[am] = -1e30f;
    }
    for (int s = 1; s < SEQ2; ++s) {           // sort ascending
        int v = ch[s], t2 = s - 1;
        while (t2 >= 0 && ch[t2] > v) { ch[t2+1] = ch[t2]; --t2; }
        ch[t2+1] = v;
    }
    for (int s = 0; s < SEQ2; ++s) idx[b * SEQ2 + s] = ch[s];
}

__global__ __launch_bounds__(256) void gather_k(const float* h, const int* idx,
                                                const float* ptok, float* out)
{
    int t = blockIdx.x * blockDim.x + threadIdx.x;
    if (t >= BZ * SEQ2 * JJ * DM) return;
    const int c = t & 255; const int r = t >> 8;
    const int j = r % JJ; const int tt = (r / JJ) % SEQ2; const int b = r / (JJ * SEQ2);
    const int f = idx[b * SEQ2 + tt];
    out[t] = h[(((size_t)b * FF + f) * JJ + j) * DM + c] + ptok[tt * DM + c];
}

// ---------------- permutes ----------------
__global__ __launch_bounds__(256) void perm_ht_k(const float* h, float* hT)
{
    int t = blockIdx.x * blockDim.x + threadIdx.x;
    if (t >= BZ * JJ * SEQ2 * DM) return;
    const int c = t & 255; const int r = t >> 8;
    const int tok = r % SEQ2; const int bj = r / SEQ2;
    const int b = bj / JJ, j = bj % JJ;
    hT[t] = h[(((size_t)b * SEQ2 + tok) * JJ + j) * DM + c];
}

__global__ __launch_bounds__(256) void perm_o_k(const float* o, float* out)
{
    int t = blockIdx.x * blockDim.x + threadIdx.x;
    if (t >= BZ * FF * JJ * DM) return;
    const int c = t & 255; const int r = t >> 8;
    const int j = r % JJ; const int f = (r / JJ) % FF; const int b = r / (JJ * FF);
    out[t] = o[(((size_t)(b * JJ + j)) * FF + f) * DM + c];
}

// ---------------- host helpers ----------------
static void launch_gemm(const float* A, const float* W, const float* bias, const float* res,
                        float* C, int M, int N, int K, int act, int NT,
                        __bf16* wscr, hipStream_t s)
{
    wprep_k<<<CDIV(K * N, 256), 256, 0, s>>>(W, wscr, K, N);
    const int tilesM = CDIV(M, 16);
    if (NT == 4) {
        const int tiles = tilesM * CDIV(N, 64);
        wmma_gemm_k<4><<<CDIV(tiles, 8), 256, 0, s>>>(A, wscr, bias, res, C, M, N, K, act);
    } else {
        const int tiles = tilesM * CDIV(N, 16);
        wmma_gemm_k<1><<<CDIV(tiles, 8), 256, 0, s>>>(A, wscr, bias, res, C, M, N, K, act);
    }
}

static void launch_ln(const float* x, const float* w, const float* b, float* y,
                      int rows, hipStream_t s)
{
    ln256_k<<<CDIV(rows, 8), 256, 0, s>>>(x, w, b, y, rows);
}

// block param offsets
enum { N1S_W=0,N1S_B,AS_QKV_W,AS_QKV_B,AS_PROJ_W,AS_PROJ_B,N2S_W,N2S_B,
       MS_W1,MS_B1,MS_W2,MS_B2,N1T_W,N1T_B,AT_QKV_W,AT_QKV_B,AT_PROJ_W,AT_PROJ_B,
       N2T_W,N2T_B,MT_W1,MT_B1,MT_W2,MT_B2 };

extern "C" void kernel_launch(void* const* d_in, const int* in_sizes, int n_in,
                              void* d_out, int out_size, void* d_ws, size_t ws_size,
                              hipStream_t stream)
{
    const int M1 = BZ * FF * JJ;          // 33048
    const int M2 = BZ * SEQ2 * JJ;        // 11016
    const size_t CT = (size_t)M2 * DM;    // cross tensor size

    const float* X      = (const float*)d_in[0];
    const float* JE_W   = (const float*)d_in[1];
    const float* JE_B   = (const float*)d_in[2];
    const float* POS    = (const float*)d_in[3];
    const float* TEMP   = (const float*)d_in[4];
    const float* PTOK   = (const float*)d_in[5];
    const float* XTOK   = (const float*)d_in[6];
    auto bp = [&](int blk, int off) { return (const float*)d_in[7 + blk * 24 + off]; };
    const float* CQ_W = (const float*)d_in[127]; const float* CQ_B = (const float*)d_in[128];
    const float* CK_W = (const float*)d_in[129]; const float* CK_B = (const float*)d_in[130];
    const float* CV_W = (const float*)d_in[131]; const float* CV_B = (const float*)d_in[132];
    const float* CP_W = (const float*)d_in[133]; const float* CP_B = (const float*)d_in[134];
    const float* NM_W = (const float*)d_in[135]; const float* NM_B = (const float*)d_in[136];
    const float* PL_W = (const float*)d_in[137]; const float* PL_B = (const float*)d_in[138];
    const float* HD_W = (const float*)d_in[139]; const float* HD_B = (const float*)d_in[140];

    // ---------------- workspace layout (floats) ----------------
    float* ws = (float*)d_ws;
    float* W0 = ws;                                 // qkv / cross o-proj      [M1*768]
    float* W1 = W0 + (size_t)M1 * 768;              // mlp hidden / cross bufs [M1*1024]
    float* W2 = W1 + (size_t)M1 * 1024;             // h                       [M1*256]
    float* W3 = W2 + (size_t)M1 * DM;               // ln / permute tmp        [M1*256]
    float* W4 = W3 + (size_t)M1 * DM;               // attn out                [M1*256]
    float* W5 = W4 + (size_t)M1 * DM;               // cluster scratch
    float* hfb   = W5;
    float* dist  = hfb   + (size_t)BZ * FF * DM;
    float* dens  = dist  + (size_t)BZ * FF * FF;
    float* dmax  = dens  + (size_t)BZ * FF;
    float* score = dmax  + BZ;
    float* qsm   = score + (size_t)BZ * FF;
    int*   idx   = (int*)(qsm + (size_t)FF * DM);
    __bf16* wscr = (__bf16*)(idx + BZ * SEQ2);      // bf16 weight scratch [<=1024*256]
    // cross-stage aliases inside W1 (mlp hidden is dead by then)
    float* hT   = W1;
    float* kbuf = W1 + CT;
    float* vbuf = W1 + 2 * CT;
    float* xat  = W1 + 3 * CT;                      // [B*J*F, 256]

    // 1. joints embed + positional/temporal embeds
    embed_k<<<CDIV(M1 * DM, 256), 256, 0, stream>>>(X, JE_W, JE_B, POS, TEMP, W2, M1 * DM);

    // 2. transformer blocks (frame pruning after block 1)
    int seq = FF;
    for (int i = 0; i < 5; ++i) {
        const int M = BZ * seq * JJ;
        // --- spatial attention ---
        launch_ln(W2, bp(i,N1S_W), bp(i,N1S_B), W3, M, stream);
        launch_gemm(W3, bp(i,AS_QKV_W), bp(i,AS_QKV_B), nullptr, W0, M, 768, 256, 0, 4, wscr, stream);
        attn_sp_k<<<CDIV(BZ*seq*HN*JJ, 128), 128, 0, stream>>>(W0, W4, BZ * seq);
        launch_gemm(W4, bp(i,AS_PROJ_W), bp(i,AS_PROJ_B), W2, W2, M, 256, 256, 0, 4, wscr, stream);
        // --- spatial MLP ---
        launch_ln(W2, bp(i,N2S_W), bp(i,N2S_B), W3, M, stream);
        launch_gemm(W3, bp(i,MS_W1), bp(i,MS_B1), nullptr, W1, M, HID, 256, 1, 4, wscr, stream);
        launch_gemm(W1, bp(i,MS_W2), bp(i,MS_B2), W2, W2, M, 256, HID, 0, 4, wscr, stream);
        // --- temporal attention ---
        launch_ln(W2, bp(i,N1T_W), bp(i,N1T_B), W3, M, stream);
        launch_gemm(W3, bp(i,AT_QKV_W), bp(i,AT_QKV_B), nullptr, W0, M, 768, 256, 0, 4, wscr, stream);
        attn_tm_k<<<CDIV(BZ*HN*JJ*seq, 128), 128, 0, stream>>>(W0, W4, seq);
        launch_gemm(W4, bp(i,AT_PROJ_W), bp(i,AT_PROJ_B), W2, W2, M, 256, 256, 0, 4, wscr, stream);
        // --- temporal MLP ---
        launch_ln(W2, bp(i,N2T_W), bp(i,N2T_B), W3, M, stream);
        launch_gemm(W3, bp(i,MT_W1), bp(i,MT_B1), nullptr, W1, M, HID, 256, 1, 4, wscr, stream);
        launch_gemm(W1, bp(i,MT_W2), bp(i,MT_B2), W2, W2, M, 256, HID, 0, 4, wscr, stream);

        if (i == 1) {   // DPC-KNN frame pruning 243 -> 81
            fmean_k   <<<CDIV(BZ*FF*DM, 256), 256, 0, stream>>>(W2, hfb);
            dist_k    <<<CDIV(BZ*FF*FF, 256), 256, 0, stream>>>(hfb, dist);
            dmax_k    <<<BZ, 256, 0, stream>>>(dist, dmax);
            density_k <<<CDIV(BZ*FF, 256), 256, 0, stream>>>(dist, dens);
            score_k   <<<CDIV(BZ*FF, 256), 256, 0, stream>>>(dist, dens, dmax, score);
            select81_k<<<BZ, 32, 0, stream>>>(score, idx);
            gather_k  <<<CDIV(M2*DM, 256), 256, 0, stream>>>(W2, idx, PTOK, W3);
            hipMemcpyAsync(W2, W3, (size_t)M2 * DM * sizeof(float),
                           hipMemcpyDeviceToDevice, stream);
            seq = SEQ2;
        }
    }

    // 3. token recovery via cross attention
    perm_ht_k<<<CDIV(M2*DM, 256), 256, 0, stream>>>(W2, hT);                 // [B*J, 81, D]
    launch_gemm(hT,   CK_W, CK_B, nullptr, kbuf, M2, 256, 256, 0, 4, wscr, stream);
    launch_gemm(hT,   CV_W, CV_B, nullptr, vbuf, M2, 256, 256, 0, 4, wscr, stream);
    launch_gemm(XTOK, CQ_W, CQ_B, nullptr, qsm,  FF, 256, 256, 0, 4, wscr, stream); // q same for all (b,j)
    attn_cr_k<<<CDIV(BZ*JJ*HN*FF, 128), 128, 0, stream>>>(qsm, kbuf, vbuf, xat);
    launch_gemm(xat, CP_W, CP_B, nullptr, W0, M1, 256, 256, 0, 4, wscr, stream);   // o-proj [B*J*F, D]
    launch_ln(W0, NM_W, NM_B, W2, M1, stream);
    perm_o_k<<<CDIV(M1*DM, 256), 256, 0, stream>>>(W2, W3);                  // -> [B,F,J,D]

    // 4. pre_logits (tanh) + head
    launch_gemm(W3, PL_W, PL_B, nullptr, W1, M1, DREP, 256, 2, 4, wscr, stream);
    launch_gemm(W1, HD_W, HD_B, nullptr, (float*)d_out, M1, 3, DREP, 0, 1, wscr, stream);
}
